// MoELLamaMLP_22943715295476
// MI455X (gfx1250) — matrix operand order repaired and checked by reading, not compiled, not run on hardware
//
#include <hip/hip_runtime.h>

// Problem constants (B=2, N=1024 -> T=2048 tokens)
#define T_TOK 2048
#define DDIM  1024
#define FDIM  2816
#define NEXP  8
#define TOPK  2

typedef __attribute__((ext_vector_type(16))) __bf16 v16bf;
typedef __attribute__((ext_vector_type(8)))  __bf16 v8bf;
typedef __attribute__((ext_vector_type(8)))  float  v8f;

// ---------------- fragment load helpers ----------------

// A-fragment (16-bit 16x32): lane (l16 = row M) holds K = {o0..o0+7, o1..o1+7}
__device__ __forceinline__ v16bf load_a_f32(const float* __restrict__ p, int o0, int o1) {
  const float4* q0 = reinterpret_cast<const float4*>(p + o0);
  const float4* q1 = reinterpret_cast<const float4*>(p + o1);
  float4 a = q0[0], b = q0[1], c = q1[0], d = q1[1];
  v16bf r;
  r[0]  = (__bf16)a.x; r[1]  = (__bf16)a.y; r[2]  = (__bf16)a.z; r[3]  = (__bf16)a.w;
  r[4]  = (__bf16)b.x; r[5]  = (__bf16)b.y; r[6]  = (__bf16)b.z; r[7]  = (__bf16)b.w;
  r[8]  = (__bf16)c.x; r[9]  = (__bf16)c.y; r[10] = (__bf16)c.z; r[11] = (__bf16)c.w;
  r[12] = (__bf16)d.x; r[13] = (__bf16)d.y; r[14] = (__bf16)d.z; r[15] = (__bf16)d.w;
  return r;
}

// B-fragment (16-bit 32x16): lane (l16 = col N) holds K = half*16 .. half*16+15 contiguous
__device__ __forceinline__ v16bf load_b_f32(const float* __restrict__ p) {
  const float4* q = reinterpret_cast<const float4*>(p);
  float4 a = q[0], b = q[1], c = q[2], d = q[3];
  v16bf r;
  r[0]  = (__bf16)a.x; r[1]  = (__bf16)a.y; r[2]  = (__bf16)a.z; r[3]  = (__bf16)a.w;
  r[4]  = (__bf16)b.x; r[5]  = (__bf16)b.y; r[6]  = (__bf16)b.z; r[7]  = (__bf16)b.w;
  r[8]  = (__bf16)c.x; r[9]  = (__bf16)c.y; r[10] = (__bf16)c.z; r[11] = (__bf16)c.w;
  r[12] = (__bf16)d.x; r[13] = (__bf16)d.y; r[14] = (__bf16)d.z; r[15] = (__bf16)d.w;
  return r;
}

__device__ __forceinline__ v16bf load_a_bf16(const __bf16* __restrict__ p, int o0, int o1) {
  v8bf g0 = *reinterpret_cast<const v8bf*>(p + o0);
  v8bf g1 = *reinterpret_cast<const v8bf*>(p + o1);
  v16bf r;
#pragma unroll
  for (int j = 0; j < 8; ++j) { r[j] = g0[j]; r[8 + j] = g1[j]; }
  return r;
}

__device__ __forceinline__ float fast_silu_mul(float g, float u) {
  // silu(g) * u with v_rcp_f32 instead of IEEE divide
  float s = __builtin_amdgcn_rcpf(1.f + __expf(-g));
  return g * s * u;
}

// ---------------- kernels ----------------

__global__ void init_k(int* counts) {
  if (threadIdx.x < NEXP) counts[threadIdx.x] = 0;
}

// One workgroup (256 threads) per token: logits -> top2 -> softmax -> counts
__global__ __launch_bounds__(256) void router_k(const float* __restrict__ x,
                                                const float* __restrict__ sw,
                                                const float* __restrict__ sb,
                                                int* __restrict__ counts,
                                                int* __restrict__ esel,
                                                float* __restrict__ wsel) {
  int t   = blockIdx.x;
  int tid = threadIdx.x;
  const float* xr = x + (size_t)t * DDIM;
  float p[NEXP];
#pragma unroll
  for (int e = 0; e < NEXP; ++e) p[e] = 0.f;
  for (int d = tid; d < DDIM; d += 256) {
    float xv = xr[d];
#pragma unroll
    for (int e = 0; e < NEXP; ++e) p[e] += xv * sw[e * DDIM + d];
  }
#pragma unroll
  for (int e = 0; e < NEXP; ++e)
    for (int off = 16; off > 0; off >>= 1) p[e] += __shfl_xor(p[e], off, 32);

  __shared__ float red[8][NEXP];
  int wave = tid >> 5, lane = tid & 31;
  if (lane == 0) {
#pragma unroll
    for (int e = 0; e < NEXP; ++e) red[wave][e] = p[e];
  }
  __syncthreads();
  if (tid == 0) {
    float l[NEXP];
#pragma unroll
    for (int e = 0; e < NEXP; ++e) {
      float acc = sb[e];
#pragma unroll
      for (int w = 0; w < 8; ++w) acc += red[w][e];
      l[e] = acc;
    }
    int e0 = 0;
#pragma unroll
    for (int e = 1; e < NEXP; ++e) if (l[e] > l[e0]) e0 = e;
    int e1 = -1;
#pragma unroll
    for (int e = 0; e < NEXP; ++e)
      if (e != e0 && (e1 < 0 || l[e] > l[e1])) e1 = e;
    float m  = l[e0];
    float a  = __expf(l[e0] - m);
    float b2 = __expf(l[e1] - m);
    float inv = __builtin_amdgcn_rcpf(a + b2);
    esel[t * 2 + 0] = e0; esel[t * 2 + 1] = e1;
    wsel[t * 2 + 0] = a * inv; wsel[t * 2 + 1] = b2 * inv;
    atomicAdd(&counts[e0], 1);
    atomicAdd(&counts[e1], 1);
  }
}

__global__ void scan_k(const int* __restrict__ counts, int* __restrict__ base,
                       int* __restrict__ cursor) {
  if (threadIdx.x == 0 && blockIdx.x == 0) {
    int acc = 0;
    for (int e = 0; e < NEXP; ++e) { base[e] = acc; acc += counts[e]; cursor[e] = 0; }
  }
}

__global__ __launch_bounds__(256) void scatter_k(const int* __restrict__ esel,
                                                 const float* __restrict__ wsel,
                                                 const int* __restrict__ base,
                                                 int* __restrict__ cursor,
                                                 int* __restrict__ tok,
                                                 int* __restrict__ slot,
                                                 float* __restrict__ wt) {
  int t = blockIdx.x * blockDim.x + threadIdx.x;
  if (t >= T_TOK) return;
#pragma unroll
  for (int k = 0; k < TOPK; ++k) {
    int e   = esel[t * 2 + k];
    int idx = base[e] + atomicAdd(&cursor[e], 1);
    tok[idx]  = t;
    slot[idx] = k;
    wt[idx]   = wsel[t * 2 + k];
  }
}

// h = silu(x@gate^T) * (x@up^T). Workgroup tile: 16 tokens x 256 F-cols.
// Each wave owns two 16-col groups (f0a, f0a+128): one A frag feeds 4 WMMAs.
__global__ __launch_bounds__(256) void gateup_k(const float* __restrict__ x,
                                                const float* __restrict__ gw,
                                                const float* __restrict__ uw,
                                                const int* __restrict__ counts,
                                                const int* __restrict__ base,
                                                const int* __restrict__ tok,
                                                __bf16* __restrict__ H) {
  int e   = blockIdx.z;
  int cnt = counts[e];
  int m0  = blockIdx.y * 16;
  if (m0 >= cnt) return;
  int beg  = base[e];
  int wave = threadIdx.x >> 5;
  int lane = threadIdx.x & 31;
  int half = lane >> 4;
  int l16  = lane & 15;
  int fA = blockIdx.x * 256 + wave * 16 + l16;
  int fB = fA + 128;
  int m = m0 + l16;
  int mm = (m < cnt) ? m : (cnt - 1);
  int tokid = tok[beg + mm];
  const float* xr  = x  + (size_t)tokid * DDIM;
  const float* grA = gw + ((size_t)e * FDIM + fA) * DDIM;
  const float* urA = uw + ((size_t)e * FDIM + fA) * DDIM;
  const float* grB = gw + ((size_t)e * FDIM + fB) * DDIM;
  const float* urB = uw + ((size_t)e * FDIM + fB) * DDIM;
  int o0 = half * 16;       // A K-group 0
  int o1 = half * 16 + 8;   // A K-group 1
  int bo = half * 16;       // B contiguous K base
  v8f cgA = {}, cuA = {}, cgB = {}, cuB = {};
  for (int d0 = 0; d0 < DDIM; d0 += 32) {
    if (d0 + 32 < DDIM) {
      __builtin_prefetch(grA + d0 + 32, 0, 3);
      __builtin_prefetch(urA + d0 + 32, 0, 3);
      __builtin_prefetch(grB + d0 + 32, 0, 3);
      __builtin_prefetch(urB + d0 + 32, 0, 3);
    }
    v16bf a = load_a_f32(xr + d0, o0, o1);
    v16bf b0 = load_b_f32(grA + d0 + bo);
    cgA = __builtin_amdgcn_wmma_f32_16x16x32_bf16(false, a, false, b0, (short)0, cgA, false, false);
    v16bf b1 = load_b_f32(urA + d0 + bo);
    cuA = __builtin_amdgcn_wmma_f32_16x16x32_bf16(false, a, false, b1, (short)0, cuA, false, false);
    v16bf b2 = load_b_f32(grB + d0 + bo);
    cgB = __builtin_amdgcn_wmma_f32_16x16x32_bf16(false, a, false, b2, (short)0, cgB, false, false);
    v16bf b3 = load_b_f32(urB + d0 + bo);
    cuB = __builtin_amdgcn_wmma_f32_16x16x32_bf16(false, a, false, b3, (short)0, cuB, false, false);
  }
#pragma unroll
  for (int r = 0; r < 8; ++r) {
    int mr = m0 + r + half * 8;
    if (mr < cnt) {
      size_t row = (size_t)(beg + mr) * FDIM;
      H[row + fA] = (__bf16)fast_silu_mul(cgA[r], cuA[r]);
      H[row + fB] = (__bf16)fast_silu_mul(cgB[r], cuB[r]);
    }
  }
}

// out_row(d) = sum_f h(f) * down_w[e][d][f], scaled by routing weight.
// Workgroup tile: 16 rows x 256 D-cols; each wave owns two 16-col groups.
__global__ __launch_bounds__(256) void down_k(const __bf16* __restrict__ H,
                                              const float* __restrict__ dw,
                                              const int* __restrict__ counts,
                                              const int* __restrict__ base,
                                              const int* __restrict__ tok,
                                              const int* __restrict__ slot,
                                              const float* __restrict__ wt,
                                              float* __restrict__ obuf) {
  int e   = blockIdx.z;
  int cnt = counts[e];
  int m0  = blockIdx.y * 16;
  if (m0 >= cnt) return;
  int beg  = base[e];
  int wave = threadIdx.x >> 5;
  int lane = threadIdx.x & 31;
  int half = lane >> 4;
  int l16  = lane & 15;
  int dA = blockIdx.x * 256 + wave * 16 + l16;
  int dB = dA + 128;
  int m = m0 + l16;
  int mm = (m < cnt) ? m : (cnt - 1);
  const __bf16* hr  = H  + (size_t)(beg + mm) * FDIM;
  const float*  wrA = dw + ((size_t)e * DDIM + dA) * FDIM;
  const float*  wrB = dw + ((size_t)e * DDIM + dB) * FDIM;
  int o0 = half * 16;
  int o1 = half * 16 + 8;
  int bo = half * 16;
  v8f cA = {}, cB = {};
  for (int f0 = 0; f0 < FDIM; f0 += 32) {
    if (f0 + 32 < FDIM) {
      __builtin_prefetch(wrA + f0 + 32, 0, 3);
      __builtin_prefetch(wrB + f0 + 32, 0, 3);
    }
    v16bf a = load_a_bf16(hr + f0, o0, o1);
    v16bf b0 = load_b_f32(wrA + f0 + bo);
    cA = __builtin_amdgcn_wmma_f32_16x16x32_bf16(false, a, false, b0, (short)0, cA, false, false);
    v16bf b1 = load_b_f32(wrB + f0 + bo);
    cB = __builtin_amdgcn_wmma_f32_16x16x32_bf16(false, a, false, b1, (short)0, cB, false, false);
  }
#pragma unroll
  for (int r = 0; r < 8; ++r) {
    int mr = m0 + r + half * 8;
    if (mr < cnt) {
      int idx = beg + mr;
      float w = wt[idx];
      int t = tok[idx], s = slot[idx];
      float* orow = obuf + ((size_t)t * TOPK + s) * DDIM;
      orow[dA] = cA[r] * w;
      orow[dB] = cB[r] * w;
    }
  }
}

__global__ __launch_bounds__(256) void combine_k(const float* __restrict__ obuf,
                                                 float* __restrict__ out) {
  size_t i = (size_t)blockIdx.x * 256 + threadIdx.x;
  if (i >= (size_t)T_TOK * DDIM) return;
  size_t t = i / DDIM, dd = i % DDIM;
  out[i] = obuf[(t * TOPK + 0) * DDIM + dd] + obuf[(t * TOPK + 1) * DDIM + dd];
}

// ---------------- launcher ----------------

extern "C" void kernel_launch(void* const* d_in, const int* in_sizes, int n_in,
                              void* d_out, int out_size, void* d_ws, size_t ws_size,
                              hipStream_t stream) {
  const float* x  = (const float*)d_in[0];
  const float* sw = (const float*)d_in[1];
  const float* sb = (const float*)d_in[2];
  const float* gw = (const float*)d_in[3];
  const float* uw = (const float*)d_in[4];
  const float* dw = (const float*)d_in[5];
  float* out = (float*)d_out;

  // carve workspace (256B aligned chunks)
  char* p = (char*)d_ws;
  auto alloc = [&](size_t bytes) -> void* {
    void* r = (void*)p;
    p += (bytes + 255) & ~(size_t)255;
    return r;
  };
  int*    counts = (int*)alloc(NEXP * sizeof(int));
  int*    cursor = (int*)alloc(NEXP * sizeof(int));
  int*    basep  = (int*)alloc(NEXP * sizeof(int));
  int*    esel   = (int*)alloc((size_t)T_TOK * TOPK * sizeof(int));
  float*  wsel   = (float*)alloc((size_t)T_TOK * TOPK * sizeof(float));
  int*    tok    = (int*)alloc((size_t)T_TOK * TOPK * sizeof(int));
  int*    slot   = (int*)alloc((size_t)T_TOK * TOPK * sizeof(int));
  float*  wt     = (float*)alloc((size_t)T_TOK * TOPK * sizeof(float));
  __bf16* H      = (__bf16*)alloc((size_t)T_TOK * TOPK * FDIM * sizeof(__bf16)); // 23 MB
  float*  obuf   = (float*)alloc((size_t)T_TOK * TOPK * DDIM * sizeof(float));   // 16.8 MB

  init_k<<<1, 32, 0, stream>>>(counts);
  router_k<<<T_TOK, 256, 0, stream>>>(x, sw, sb, counts, esel, wsel);
  scan_k<<<1, 32, 0, stream>>>(counts, basep, cursor);
  scatter_k<<<(T_TOK + 255) / 256, 256, 0, stream>>>(esel, wsel, basep, cursor, tok, slot, wt);
  gateup_k<<<dim3(FDIM / 256, T_TOK / 16, NEXP), 256, 0, stream>>>(x, gw, uw, counts, basep, tok, H);
  down_k<<<dim3(DDIM / 256, T_TOK / 16, NEXP), 256, 0, stream>>>(H, dw, counts, basep, tok, slot, wt, obuf);
  combine_k<<<((size_t)T_TOK * DDIM + 255) / 256, 256, 0, stream>>>(obuf, out);
}
